// VectorQuantizer_3865470566539
// MI455X (gfx1250) — matrix operand order repaired and checked by reading, not compiled, not run on hardware
//
#include <hip/hip_runtime.h>
#include <hip/hip_bf16.h>

typedef __attribute__((ext_vector_type(16))) __bf16 v16bf;
typedef __attribute__((ext_vector_type(8)))  __bf16 v8bf;
typedef __attribute__((ext_vector_type(8)))  float  v8f;

#define D      256
#define KCB    1024
#define NROWS  32768
#define MT     128            // latent rows per block
#define NT     128            // codebook entries per LDS chunk
#define LSTR   264            // padded bf16 row stride (bank spread)

// workspace layout (float units)
#define WS_NORM 0             // [0..1023]   codebook squared norms (f32)
#define WS_INDS 1024          // [1024..]    32768 int32 argmin indices
#define WS_PART 33792         // [33792..]   8192 per-block loss partials
#define NPART   8192

#define QOFF  ((size_t)NROWS * D)   // 8388608: vq_loss at QOFF, inds at QOFF+1

// ---------------------------------------------------------------- norms ----
__global__ void vq_norms(const float* __restrict__ cb, float* __restrict__ ws) {
  const int lane = threadIdx.x & 31;
  const int wv   = threadIdx.x >> 5;
  const int row  = blockIdx.x * 8 + wv;           // 128 blocks x 8 waves = 1024
  const float4* r4 = (const float4*)(cb + (size_t)row * D);
  float4 a = r4[lane * 2 + 0];
  float4 b = r4[lane * 2 + 1];
  float s = a.x*a.x + a.y*a.y + a.z*a.z + a.w*a.w
          + b.x*b.x + b.y*b.y + b.z*b.z + b.w*b.w;
  #pragma unroll
  for (int off = 16; off >= 1; off >>= 1) s += __shfl_xor(s, off, 32);
  if (lane == 0) ws[WS_NORM + row] = s;
}

// ------------------------------------------------------- GEMM + argmin ----
__device__ __forceinline__ v16bf load_bfrag(const __bf16* p) {
  v8bf b0 = *(const v8bf*)p;
  v8bf b1 = *(const v8bf*)(p + 8);
  return __builtin_shufflevector(b0, b1, 0,1,2,3,4,5,6,7,8,9,10,11,12,13,14,15);
}

__global__ void __launch_bounds__(256) vq_argmin(const float* __restrict__ X,
                                                 const float* __restrict__ E,
                                                 const float* __restrict__ ws,
                                                 int* __restrict__ inds) {
  __shared__ __bf16 As[MT * LSTR];   // 67.6 KB
  __shared__ __bf16 Bs[NT * LSTR];   // 67.6 KB
  __shared__ float  nrm[NT];

  const int tid  = threadIdx.x;
  const int lane = tid & 31;
  const int wv   = tid >> 5;
  const int h    = lane >> 4;       // half-wave select
  const int l16  = lane & 15;
  const int m0   = blockIdx.x * MT;

  // ---- Stage A: latent tile f32 -> bf16 LDS (16 iters x 256 thr x 8 elems)
  #pragma unroll
  for (int it = 0; it < (MT * D / 8) / 256; ++it) {
    int c8  = it * 256 + tid;
    int r   = c8 >> 5;              // 32 x 8-elem chunks per row
    int col = (c8 & 31) * 8;
    const float4* p = (const float4*)(X + (size_t)(m0 + r) * D + col);
    float4 u = p[0], v = p[1];
    v8bf w;
    w[0] = (__bf16)u.x; w[1] = (__bf16)u.y; w[2] = (__bf16)u.z; w[3] = (__bf16)u.w;
    w[4] = (__bf16)v.x; w[5] = (__bf16)v.y; w[6] = (__bf16)v.z; w[7] = (__bf16)v.w;
    *(v8bf*)&As[r * LSTR + col] = w;
  }
  __syncthreads();

  // ---- A fragments: one 16-row strip per wave, all of K=256 in VGPRs.
  // A layout (matches B): lanes 0-15 hold K=0..15; lanes 16-31 hold
  // K=16..31  ->  lo @ d0+h*16, hi @ d0+h*16+8.
  v16bf afr[8];
  {
    const __bf16* arow = &As[(wv * 16 + l16) * LSTR];
    #pragma unroll
    for (int kc = 0; kc < 8; ++kc) {
      v8bf lo = *(const v8bf*)(arow + kc * 32 + h * 16);
      v8bf hi = *(const v8bf*)(arow + kc * 32 + h * 16 + 8);
      afr[kc] = __builtin_shufflevector(lo, hi, 0,1,2,3,4,5,6,7,8,9,10,11,12,13,14,15);
    }
  }

  float minv[8];
  int   mini[8];
  #pragma unroll
  for (int r = 0; r < 8; ++r) { minv[r] = 3.402823466e38f; mini[r] = 0; }

  for (int n0 = 0; n0 < KCB; n0 += NT) {
    __syncthreads();
    // ---- Stage B: codebook tile f32 -> bf16 LDS
    #pragma unroll
    for (int it = 0; it < (NT * D / 8) / 256; ++it) {
      int c8  = it * 256 + tid;
      int r   = c8 >> 5;
      int col = (c8 & 31) * 8;
      const float4* p = (const float4*)(E + (size_t)(n0 + r) * D + col);
      float4 u = p[0], v = p[1];
      v8bf w;
      w[0] = (__bf16)u.x; w[1] = (__bf16)u.y; w[2] = (__bf16)u.z; w[3] = (__bf16)u.w;
      w[4] = (__bf16)v.x; w[5] = (__bf16)v.y; w[6] = (__bf16)v.z; w[7] = (__bf16)v.w;
      *(v8bf*)&Bs[r * LSTR + col] = w;
    }
    if (tid < NT) nrm[tid] = ws[WS_NORM + n0 + tid];
    __syncthreads();

    // ---- Two 16-entry subtiles at a time: two independent WMMA accumulator
    // chains sharing each A fragment, with double-buffered B fragments so the
    // ds_load_b128 pair for kc+1 overlaps the WMMAs on kc.
    // B layout used here (contiguous, matches A): lanes 0-15 hold K=0..15, lanes 16-31 hold K=16..31
    // -> 16 contiguous bf16 at d0 + h*16 of column n = l16.
    #pragma unroll
    for (int ns = 0; ns < NT / 16; ns += 2) {
      const __bf16* br0 = &Bs[(ns * 16 + l16) * LSTR + h * 16];
      const __bf16* br1 = br0 + 16 * LSTR;
      v8f acc0 = {};
      v8f acc1 = {};
      v16bf b0 = load_bfrag(br0);
      v16bf b1 = load_bfrag(br1);
      #pragma unroll
      for (int kc = 0; kc < 8; ++kc) {
        v16bf bn0, bn1;
        if (kc < 7) {
          bn0 = load_bfrag(br0 + (kc + 1) * 32);   // prefetch next K-chunk
          bn1 = load_bfrag(br1 + (kc + 1) * 32);
        }
        acc0 = __builtin_amdgcn_wmma_f32_16x16x32_bf16(
            false, afr[kc], false, b0, (short)0, acc0, false, false);
        acc1 = __builtin_amdgcn_wmma_f32_16x16x32_bf16(
            false, afr[kc], false, b1, (short)0, acc1, false, false);
        if (kc < 7) { b0 = bn0; b1 = bn1; }
      }
      // comparator: ||e||^2 - 2 x.e  (||x||^2 constant per row, dropped)
      float nn0   = nrm[ns * 16 + l16];
      float nn1   = nrm[ns * 16 + 16 + l16];
      int   nidx0 = n0 + ns * 16 + l16;
      int   nidx1 = nidx0 + 16;
      #pragma unroll
      for (int r = 0; r < 8; ++r) {
        float dv0 = nn0 - 2.0f * acc0[r];
        float dv1 = nn1 - 2.0f * acc1[r];
        if (dv0 < minv[r]) { minv[r] = dv0; mini[r] = nidx0; }  // ascending n:
        if (dv1 < minv[r]) { minv[r] = dv1; mini[r] = nidx1; }  // earliest wins
      }
    }
  }

  // ---- C layout: VGPR r -> row M = wv*16 + r + 8*h, col N = l16.
  // Min-reduce across the 16 lanes of each half-wave (distinct N classes),
  // tie-break on lowest index to match jnp.argmin.
  #pragma unroll
  for (int r = 0; r < 8; ++r) {
    float v  = minv[r];
    int   ix = mini[r];
    #pragma unroll
    for (int off = 8; off >= 1; off >>= 1) {
      float ov  = __shfl_xor(v, off, 16);
      int   oix = __shfl_xor(ix, off, 16);
      if (ov < v || (ov == v && oix < ix)) { v = ov; ix = oix; }
    }
    if (l16 == 0) inds[m0 + wv * 16 + r + 8 * h] = ix;
  }
}

// ------------------------------------------------- gather + loss pass ----
__global__ void vq_quantize(const float* __restrict__ X, const float* __restrict__ E,
                            const int* __restrict__ inds, float* __restrict__ out,
                            float* __restrict__ partials) {
  __shared__ float red[256];
  const int tid = threadIdx.x;
  const int rl  = tid >> 6;                 // 4 rows per block
  const int q   = tid & 63;                 // float4 lane within row (D/4=64)
  const int n   = blockIdx.x * 4 + rl;
  const int ind = inds[n];
  const float4 e = ((const float4*)(E + (size_t)ind * D))[q];
  const float4 x = ((const float4*)(X + (size_t)n * D))[q];
  ((float4*)out)[(size_t)n * 64 + q] = e;   // straight-through value == quantized
  float dx = e.x - x.x, dy = e.y - x.y, dz = e.z - x.z, dw = e.w - x.w;
  if (q == 0) out[QOFF + 1 + n] = (float)ind;
  red[tid] = dx*dx + dy*dy + dz*dz + dw*dw;
  __syncthreads();
  #pragma unroll
  for (int st = 128; st >= 1; st >>= 1) {
    if (tid < st) red[tid] += red[tid + st];
    __syncthreads();
  }
  if (tid == 0) partials[blockIdx.x] = red[0];
}

__global__ void vq_finalize(const float* __restrict__ partials, float* __restrict__ out) {
  __shared__ float red[256];
  const int tid = threadIdx.x;
  float s = 0.f;
  for (int i = tid; i < NPART; i += 256) s += partials[i];   // fixed order: deterministic
  red[tid] = s;
  __syncthreads();
  #pragma unroll
  for (int st = 128; st >= 1; st >>= 1) {
    if (tid < st) red[tid] += red[tid + st];
    __syncthreads();
  }
  // embedding_loss == commitment_loss numerically -> vq = (1 + 0.25) * mse
  if (tid == 0) out[QOFF] = 1.25f * red[0] / (float)((size_t)NROWS * D);
}

// ----------------------------------------------------------------------
extern "C" void kernel_launch(void* const* d_in, const int* in_sizes, int n_in,
                              void* d_out, int out_size, void* d_ws, size_t ws_size,
                              hipStream_t stream) {
  const float* X = (const float*)d_in[0];   // latents  [32,1024,256] f32
  const float* E = (const float*)d_in[1];   // codebook [1024,256]    f32
  float* out = (float*)d_out;               // [8388608 q | 1 loss | 32768 inds]
  float* ws  = (float*)d_ws;
  int*   inds = (int*)(ws + WS_INDS);
  float* partials = ws + WS_PART;

  vq_norms   <<<KCB / 8, 256, 0, stream>>>(E, ws);
  vq_argmin  <<<NROWS / MT, 256, 0, stream>>>(X, E, ws, inds);
  vq_quantize<<<NROWS / 4, 256, 0, stream>>>(X, E, inds, out, partials);
  vq_finalize<<<1, 256, 0, stream>>>(partials, out);
}
